// THPLayer_73212012527973
// MI455X (gfx1250) — compile-verified, matrix operand order checked by reading, not run on hardware
//
#include <hip/hip_runtime.h>
#include <hip/hip_bf16.h>
#include <math.h>

// ---------------------------------------------------------------------------
// Problem constants (reference: B=8, N=2048, D=512)
// ---------------------------------------------------------------------------
#define BATCH 8
#define NSEQ  2048
#define DMODEL 512
#define INV_TEMP 0.04419417382415922f   // 1/sqrt(512)
#define EPS 1e-5f

typedef __attribute__((ext_vector_type(16))) _Float16 v16h;
typedef __attribute__((ext_vector_type(8)))  _Float16 v8h;
typedef __attribute__((ext_vector_type(8)))  float    v8f;
typedef __attribute__((ext_vector_type(4)))  unsigned int v4u;
typedef __attribute__((ext_vector_type(8)))  int      v8i;
typedef __attribute__((ext_vector_type(4)))  int      v4i;

// ---------------------------------------------------------------------------
// 16x32 f16 A-fragment (== matching 32x16 B-fragment) from a row-major matrix.
// Per ISA 7.12.2: lane l<16: row l, K 0..7 in v0..3, K 16..23 in v4..7;
// lane l+16: row l, K 8..15 / 24..31.
// ---------------------------------------------------------------------------
__device__ __forceinline__ v16h load_frag_rm(const _Float16* base, int stride, int lane) {
    const int r  = lane & 15;
    const int hs = (lane >> 4) << 3;             // 0 or 8
    const _Float16* p = base + (size_t)r * stride + hs;
    v8h lo = *(const v8h*)(p);
    v8h hi = *(const v8h*)(p + 16);
    return __builtin_shufflevector(lo, hi, 0,1,2,3,4,5,6,7,8,9,10,11,12,13,14,15);
}

__device__ __forceinline__ v8f wmma_f16(v16h a, v16h b, v8f c) {
    return __builtin_amdgcn_wmma_f32_16x16x32_f16(
        false, a, false, b, (short)0, c, false, false);
}

// branchless ELU (alpha = 1) on the fast-exp path (v_exp_f32)
__device__ __forceinline__ float elu_fast(float x) {
    float e = __expf(x) - 1.0f;
    return (x > 0.0f) ? x : e;
}

// ---------------------------------------------------------------------------
// TDM: DMA a 128-row x 512-half (2-byte) row-major tile, global -> LDS.
// D# per CDNA5 ISA sec.8: group0 = {count=1 | lds_addr | global_addr(57b) |
// type=2}; group1 = {data_size=1(2B), tensor_dim0=512, tensor_dim1=128,
// tile_dim0=512, tile_dim1=128, dim0_stride=512}. Groups 2/3 zero (2D tensor).
// Tracked by TENSORcnt (per-wave); issue from one wave only.
// ---------------------------------------------------------------------------
__device__ __forceinline__ void tdm_load_k_tile(const _Float16* gsrc, _Float16* ldst) {
    const unsigned long long ga = (unsigned long long)(uintptr_t)gsrc;
    const unsigned int lo = (unsigned int)(uintptr_t)ldst;  // LDS byte offset
    v4u g0 = {};
    g0[0] = 1u;                                              // count=1, user D#
    g0[1] = lo;                                              // lds_addr
    g0[2] = (unsigned int)(ga & 0xffffffffu);                // global_addr[31:0]
    g0[3] = (unsigned int)((ga >> 32) & 0x01ffffffu)         // global_addr[56:32]
          | (2u << 30);                                      // type = 2 (image)
    v8i g1 = {};
    g1[0] = (int)(1u << 16);          // data_size = 1 -> 2 bytes/elem
    g1[1] = (int)(512u << 16);        // tensor_dim0[15:0]  (bits 63:48)
    g1[2] = (int)(128u << 16);        // tensor_dim1[15:0]  (bits 95:80)
    g1[3] = (int)(512u << 16);        // tile_dim0          (bits 127:112)
    g1[4] = (int)(128u);              // tile_dim1          (bits 143:128)
    g1[5] = (int)(512u);              // tensor_dim0_stride (bits 191:160)
    v4i z4 = {};
#if __clang_major__ >= 23
    v8i z8 = {};
    __builtin_amdgcn_tensor_load_to_lds(g0, g1, z4, z4, z8, 0);
#else
    __builtin_amdgcn_tensor_load_to_lds(g0, g1, z4, z4, 0);
#endif
}

// ---------------------------------------------------------------------------
// Kernel 0: f32 -> f16 conversion (X and W), 8 elems/thread
// ---------------------------------------------------------------------------
__global__ void cvt_f32_to_f16(const float* __restrict__ in,
                               _Float16* __restrict__ out, int n) {
    int i = (blockIdx.x * blockDim.x + threadIdx.x) * 8;
    if (i >= n) return;
    float4 a = *(const float4*)(in + i);
    float4 b = *(const float4*)(in + i + 4);
    v8h h;
    h[0] = (_Float16)a.x; h[1] = (_Float16)a.y;
    h[2] = (_Float16)a.z; h[3] = (_Float16)a.w;
    h[4] = (_Float16)b.x; h[5] = (_Float16)b.y;
    h[6] = (_Float16)b.z; h[7] = (_Float16)b.w;
    *(v8h*)(out + i) = h;
}

// ---------------------------------------------------------------------------
// Kernel 1: Vt[b][e][n] = elu( X W^T + b )^T  (f16, transposed layout).
// Grid: BATCH*(NSEQ/16) x 256 threads (8 waves); wave w owns e-cols [w*64,+64).
// ---------------------------------------------------------------------------
__global__ void __launch_bounds__(256)
proj_elu_kernel(const _Float16* __restrict__ Xh,
                const _Float16* __restrict__ Wh,
                const float*    __restrict__ bias,
                _Float16*       __restrict__ Vt) {
    const int bidx = blockIdx.x / (NSEQ / 16);
    const int n0   = (blockIdx.x % (NSEQ / 16)) * 16;
    const int w    = threadIdx.x >> 5;
    const int lane = threadIdx.x & 31;

    const _Float16* xrow  = Xh + ((size_t)bidx * NSEQ + n0) * DMODEL;
    const _Float16* wbase = Wh + (size_t)(w * 64) * DMODEL;

    v8f acc[4] = {};
#pragma unroll 4
    for (int kk = 0; kk < DMODEL / 32; ++kk) {
        v16h aX = load_frag_rm(xrow + kk * 32, DMODEL, lane);
#pragma unroll
        for (int j = 0; j < 4; ++j) {
            v16h bW = load_frag_rm(wbase + (size_t)(j * 16) * DMODEL + kk * 32,
                                   DMODEL, lane);
            acc[j] = wmma_f16(aX, bW, acc[j]);
        }
    }

#pragma unroll
    for (int j = 0; j < 4; ++j) {
        const int e = w * 64 + j * 16 + (lane & 15);
        const float bv = bias[e];
        v8h h;
#pragma unroll
        for (int v = 0; v < 8; ++v)
            h[v] = (_Float16)elu_fast(acc[j][v] + bv);
        _Float16* dst = Vt + ((size_t)bidx * DMODEL + e) * NSEQ
                           + n0 + ((lane >> 4) << 3);
        *(v8h*)dst = h;
    }
}

// ---------------------------------------------------------------------------
// Kernel 2: fused  S = mask .* (X Xt / sqrt(D));  out = (S / max(||S||2,eps)) V
// Grid: BATCH*(NSEQ/32) x 512 threads (16 waves); 32-row n-tile, m-steps of 128.
// TDM double-buffers the 128x512 f16 K-tile in LDS (TENSORcnt + barriers),
// so phase-1 compute of step t overlaps the DMA of step t+1.
// Dynamic LDS: sK0 128KB | sK1 128KB | sQ 32KB | sS 8KB | sumsq 128B.
// ---------------------------------------------------------------------------
#define SK_BYTES   (128 * DMODEL * 2)          // 131072
#define SQ_BYTES   (32 * DMODEL * 2)           //  32768
#define SS_BYTES   (32 * 128 * 2)              //   8192
#define SMEM_TOTAL (2 * SK_BYTES + SQ_BYTES + SS_BYTES + 128)

__global__ void __launch_bounds__(512)
fused_attn_kernel(const _Float16* __restrict__ Xh,
                  const float*    __restrict__ mask,
                  const _Float16* __restrict__ Vt,
                  float*          __restrict__ out) {
    extern __shared__ __align__(16) char smem[];
    _Float16* sK0    = (_Float16*)(smem);
    _Float16* sK1    = (_Float16*)(smem + SK_BYTES);
    _Float16* sQ     = (_Float16*)(smem + 2 * SK_BYTES);
    _Float16* sS     = (_Float16*)(smem + 2 * SK_BYTES + SQ_BYTES);
    float*    sSumSq = (float*)   (smem + 2 * SK_BYTES + SQ_BYTES + SS_BYTES);

    const int bidx = blockIdx.x / (NSEQ / 32);
    const int n0   = (blockIdx.x % (NSEQ / 32)) * 32;
    const int tid  = threadIdx.x;
    const int w    = tid >> 5;
    const int lane = tid & 31;

    const _Float16* Kg = Xh + (size_t)bidx * NSEQ * DMODEL;  // this batch's keys

    {   // stage Q tile into LDS
        const v8h* src = (const v8h*)(Xh + ((size_t)bidx * NSEQ + n0) * DMODEL);
        v8h* dst = (v8h*)sQ;
#pragma unroll
        for (int i = 0; i < 4; ++i)
            dst[tid + i * 512] = src[tid + i * 512];
    }
    if (tid < 32) sSumSq[tid] = 0.0f;

    // prime the K pipeline: DMA tile 0 into sK0 (wave 0 only; TDM ignores EXEC)
    if (w == 0) {
        tdm_load_k_tile(Kg, sK0);
        __builtin_amdgcn_s_wait_tensorcnt(0);
    }
    __syncthreads();

    const int rt = w & 1;           // score / out row-tile
    const int ct = w >> 1;          // phase-1 col-tile (0..7)
    const int eg = w >> 1;          // phase-2 e-group

    v8f oacc[4] = {};

    for (int t = 0; t < NSEQ / 128; ++t) {
        const int mstep = t * 128;
        _Float16* sKcur = (t & 1) ? sK1 : sK0;

        // kick off the next tile's DMA while we compute on the current one
        if (w == 0 && t + 1 < NSEQ / 128)
            tdm_load_k_tile(Kg + (size_t)(t + 1) * 128 * DMODEL,
                            (t & 1) ? sK0 : sK1);

        // ================= phase 1: masked scores (Q,K both from LDS) ====
        {
            const _Float16* qbase = sQ + (size_t)(rt * 16) * DMODEL;
            const _Float16* kbase = sKcur + (size_t)(ct * 16) * DMODEL;

            v8f s = {};
#pragma unroll 4
            for (int kk = 0; kk < DMODEL / 32; ++kk) {
                v16h aQ = load_frag_rm(qbase + kk * 32, DMODEL, lane);
                v16h bK = load_frag_rm(kbase + kk * 32, DMODEL, lane);
                s = wmma_f16(aQ, bK, s);
            }

            const float* mp = mask
                + ((size_t)bidx * NSEQ + n0 + rt * 16 + ((lane >> 4) << 3)) * NSEQ
                + mstep + ct * 16 + (lane & 15);
            if (mstep + 128 < NSEQ)
                __builtin_prefetch(mp + 128, 0, 0);    // global_prefetch_b8
#pragma unroll
            for (int v = 0; v < 8; ++v) {
                float mv = __builtin_nontemporal_load(mp + (size_t)v * NSEQ);
                float sv = s[v] * INV_TEMP * mv;
                float sq = sv * sv;
                sq += __shfl_xor(sq, 1);
                sq += __shfl_xor(sq, 2);
                sq += __shfl_xor(sq, 4);
                sq += __shfl_xor(sq, 8);
                const int row = rt * 16 + ((lane >> 4) << 3) + v;
                if ((lane & 15) == 0) atomicAdd(&sSumSq[row], sq);
                sS[(size_t)row * 128 + ct * 16 + (lane & 15)] = (_Float16)sv;
            }
        }
        __syncthreads();

        // ================= phase 2: out += S * Vt ========================
        {
            v16h aS0 = load_frag_rm(sS + (size_t)(rt * 16) * 128 +  0, 128, lane);
            v16h aS1 = load_frag_rm(sS + (size_t)(rt * 16) * 128 + 32, 128, lane);
            v16h aS2 = load_frag_rm(sS + (size_t)(rt * 16) * 128 + 64, 128, lane);
            v16h aS3 = load_frag_rm(sS + (size_t)(rt * 16) * 128 + 96, 128, lane);
            const _Float16* vbase =
                Vt + ((size_t)bidx * DMODEL + eg * 64) * NSEQ + mstep;
#pragma unroll
            for (int j = 0; j < 4; ++j) {
                const _Float16* vj = vbase + (size_t)(j * 16) * NSEQ;
                v16h c0 = load_frag_rm(vj +  0, NSEQ, lane);
                v16h c1 = load_frag_rm(vj + 32, NSEQ, lane);
                v16h c2 = load_frag_rm(vj + 64, NSEQ, lane);
                v16h c3 = load_frag_rm(vj + 96, NSEQ, lane);
                oacc[j] = wmma_f16(aS0, c0, oacc[j]);
                oacc[j] = wmma_f16(aS1, c1, oacc[j]);
                oacc[j] = wmma_f16(aS2, c2, oacc[j]);
                oacc[j] = wmma_f16(aS3, c3, oacc[j]);
            }
        }

        // make sure the next K tile has landed before anyone reads it
        if (w == 0 && t + 1 < NSEQ / 128)
            __builtin_amdgcn_s_wait_tensorcnt(0);
        __syncthreads();
    }

    // ---- epilogue: L2-normalize rows, store ------------------------------
#pragma unroll
    for (int v = 0; v < 8; ++v) {
        const int row = rt * 16 + ((lane >> 4) << 3) + v;
        const float sc = 1.0f / fmaxf(sqrtf(sSumSq[row]), EPS);
        float* op = out + ((size_t)bidx * NSEQ + n0 + row) * DMODEL
                        + eg * 64 + (lane & 15);
#pragma unroll
        for (int j = 0; j < 4; ++j)
            op[j * 16] = oacc[j][v] * sc;
    }
}

// ---------------------------------------------------------------------------
// Launch
// ---------------------------------------------------------------------------
extern "C" void kernel_launch(void* const* d_in, const int* in_sizes, int n_in,
                              void* d_out, int out_size, void* d_ws, size_t ws_size,
                              hipStream_t stream) {
    const float* X    = (const float*)d_in[0];   // [8,2048,512]
    const float* M    = (const float*)d_in[1];   // [8,2048,2048]
    const float* W    = (const float*)d_in[2];   // [512,512]
    const float* bias = (const float*)d_in[3];   // [512]
    float* out = (float*)d_out;

    const size_t XH_ELEMS = (size_t)BATCH * NSEQ * DMODEL;  // 8,388,608
    const size_t WH_ELEMS = (size_t)DMODEL * DMODEL;        //   262,144
    char* ws = (char*)d_ws;
    _Float16* Xh = (_Float16*)ws;
    _Float16* Wh = (_Float16*)(ws + XH_ELEMS * sizeof(_Float16));
    _Float16* Vt = (_Float16*)(ws + (XH_ELEMS + WH_ELEMS) * sizeof(_Float16));

    cvt_f32_to_f16<<<(int)(XH_ELEMS / (256 * 8)), 256, 0, stream>>>(X, Xh, (int)XH_ELEMS);
    cvt_f32_to_f16<<<(int)(WH_ELEMS / (256 * 8)), 256, 0, stream>>>(W, Wh, (int)WH_ELEMS);

    proj_elu_kernel<<<BATCH * (NSEQ / 16), 256, 0, stream>>>(Xh, Wh, bias, Vt);

    fused_attn_kernel<<<BATCH * (NSEQ / 32), 512, SMEM_TOTAL, stream>>>(Xh, M, Vt, out);
}
